// TFEBTM_74380243632188
// MI455X (gfx1250) — matrix operand order repaired and checked by reading, not compiled, not run on hardware
//
#include <hip/hip_runtime.h>
#include <hip/hip_bf16.h>

// ---------------- types ----------------
typedef __attribute__((ext_vector_type(16))) _Float16 v16h;
typedef __attribute__((ext_vector_type(8)))  _Float16 v8h;
typedef __attribute__((ext_vector_type(8)))  float    v8f;

// ---------------- problem constants ----------------
#define PV 50000
#define PVPAD 50048         // V padded to multiple of 128 (M-tile)
#define PE 300
#define PH 800
#define PHPAD 832           // H padded to multiple of 64 (N-tile) for weight buffers
#define PK 128
#define PB 131072
#define KPAD_E 320          // 300 padded to multiple of 32
#define BCHUNK 16384
#define NCHUNK (PB / BCHUNK)

// A fragment 16x32 f16 from an f16 row-major matrix [Mpad x lda] (no bounds guard;
// caller guarantees padded allocation). lane<16: K = k0+{0..15};
// lane>=16: K = k0+16+{0..15}
__device__ __forceinline__ v16h load_afrag(const _Float16* __restrict__ A, int lda,
                                           int m0, int k0, int lane) {
    int row = m0 + (lane & 15);
    int hi  = lane >> 4;
    const _Float16* p = A + (size_t)row * lda + k0 + hi * 16;
    return *(const v16h*)(p);
}

// B fragment 32x16 f16 from pre-transposed weights Bt[Npad][Kpad] (row n = output col n,
// zero-padded to Npad; no bounds guard). lane<16: K = k0+{0..15}; lane>=16: K = k0+16+{0..15}
__device__ __forceinline__ v16h load_bfrag(const _Float16* __restrict__ Bt, int Kpad,
                                           int n0, int k0, int lane) {
    int col = n0 + (lane & 15);
    int hi  = lane >> 4;
    const _Float16* p = Bt + (size_t)col * Kpad + k0 + hi * 16;
    return *(const v16h*)(p);
}

#define WMMA_F16(a, b, c) \
    __builtin_amdgcn_wmma_f32_16x16x32_f16(false, (a), false, (b), (short)0, (c), false, false)

// ---------------- GEMM: D = act(A @ Bt^T + bias) ----------------
// block = 256 threads = 8 waves; wave tile = 16(M) x 64(N); block tile = 128(M) x 64(N)
// Software-pipelined: loads for k-step i+1 are issued before the WMMAs of step i.
// A has >= gridDim.x*128 rows; Bt has >= gridDim.y*64 zero-padded rows. Stores are guarded.
template <bool ACT_TANH, bool OUT_F16>
__global__ __launch_bounds__(256) void gemm_kernel(
    const _Float16* __restrict__ A, int lda,
    const _Float16* __restrict__ Bt, int Kpad,
    const float* __restrict__ bias,
    float* __restrict__ outF, _Float16* __restrict__ outH, int ldo,
    int M, int N) {
    const int lane = threadIdx.x & 31;
    const int wave = threadIdx.x >> 5;
    const int m0   = blockIdx.x * 128 + wave * 16;
    const int n0w  = blockIdx.y * 64;

    v8f acc0, acc1, acc2, acc3;
#pragma unroll
    for (int i = 0; i < 8; ++i) { acc0[i] = 0.f; acc1[i] = 0.f; acc2[i] = 0.f; acc3[i] = 0.f; }

    v16h a  = load_afrag(A, lda, m0, 0, lane);
    v16h b0 = load_bfrag(Bt, Kpad, n0w +  0, 0, lane);
    v16h b1 = load_bfrag(Bt, Kpad, n0w + 16, 0, lane);
    v16h b2 = load_bfrag(Bt, Kpad, n0w + 32, 0, lane);
    v16h b3 = load_bfrag(Bt, Kpad, n0w + 48, 0, lane);

    for (int k0 = 32; k0 < Kpad; k0 += 32) {
        v16h an  = load_afrag(A, lda, m0, k0, lane);
        v16h c0  = load_bfrag(Bt, Kpad, n0w +  0, k0, lane);
        v16h c1  = load_bfrag(Bt, Kpad, n0w + 16, k0, lane);
        v16h c2  = load_bfrag(Bt, Kpad, n0w + 32, k0, lane);
        v16h c3  = load_bfrag(Bt, Kpad, n0w + 48, k0, lane);
        acc0 = WMMA_F16(a, b0, acc0);
        acc1 = WMMA_F16(a, b1, acc1);
        acc2 = WMMA_F16(a, b2, acc2);
        acc3 = WMMA_F16(a, b3, acc3);
        a = an; b0 = c0; b1 = c1; b2 = c2; b3 = c3;
    }
    acc0 = WMMA_F16(a, b0, acc0);
    acc1 = WMMA_F16(a, b1, acc1);
    acc2 = WMMA_F16(a, b2, acc2);
    acc3 = WMMA_F16(a, b3, acc3);

    const int colBase = lane & 15;
    const int hi      = lane >> 4;
    v8f accs[4] = {acc0, acc1, acc2, acc3};
#pragma unroll
    for (int t = 0; t < 4; ++t) {
        int n = n0w + t * 16 + colBase;
        if (n >= N) continue;
        float bv = bias ? bias[n] : 0.0f;
#pragma unroll
        for (int r = 0; r < 8; ++r) {
            int m = m0 + hi * 8 + r;
            if (m < M) {
                float v = accs[t][r] + bv;
                if (ACT_TANH) v = tanhf(v);
                if (OUT_F16) outH[(size_t)m * ldo + n] = (_Float16)v;
                else         outF[(size_t)m * ldo + n] = v;
            }
        }
    }
}

// ---------------- transpose + f32->f16 convert (weights), zero-pad K and N ----------------
__global__ void prep_transpose(const float* __restrict__ in, _Float16* __restrict__ out,
                               int Kreal, int Nreal, int Kpad, int Npad) {
    int idx = blockIdx.x * blockDim.x + threadIdx.x;
    if (idx >= Npad * Kpad) return;
    int n = idx / Kpad, k = idx - n * Kpad;
    float v = (n < Nreal && k < Kreal) ? in[(size_t)k * Nreal + n] : 0.0f;
    out[idx] = (_Float16)v;
}

// ---------------- row-major f32 -> padded f16 convert (activations), zero-pad K and rows ----------------
__global__ void pad_convert(const float* __restrict__ in, _Float16* __restrict__ out,
                            int Kreal, int Kpad, long Mreal, long total) {
    long idx = (long)blockIdx.x * blockDim.x + threadIdx.x;
    if (idx >= total) return;
    long r = idx / Kpad;
    int  k = (int)(idx - r * Kpad);
    float v = (r < Mreal && k < Kreal) ? in[r * Kreal + k] : 0.0f;
    out[idx] = (_Float16)v;
}

// ---------------- column softmax stats over V for each of K columns ----------------
__global__ __launch_bounds__(256) void col_stats(const float* __restrict__ logits,
                                                 float* __restrict__ colmax,
                                                 float* __restrict__ colsum) {
    const int k = blockIdx.x;
    const int tid = threadIdx.x;
    __shared__ float red[256];
    float m = -3.4e38f;
    for (int i = tid; i < PV; i += 256) m = fmaxf(m, logits[(size_t)i * PK + k]);
    red[tid] = m; __syncthreads();
    for (int s = 128; s > 0; s >>= 1) { if (tid < s) red[tid] = fmaxf(red[tid], red[tid + s]); __syncthreads(); }
    float cm = red[0]; __syncthreads();
    float s = 0.0f;
    for (int i = tid; i < PV; i += 256) s += expf(logits[(size_t)i * PK + k] - cm);
    red[tid] = s; __syncthreads();
    for (int st = 128; st > 0; st >>= 1) { if (tid < st) red[tid] += red[tid + st]; __syncthreads(); }
    if (tid == 0) { colmax[k] = cm; colsum[k] = red[0]; }
}

// ---------------- per-row finalize: softmax(mu), kld terms, tb gather, log(res) ----------------
// one wave per row; 8 rows per block; writes per-block partial (recon, kld) pairs
__global__ __launch_bounds__(256) void finalize_rows(
    const float* __restrict__ muc, const float* __restrict__ lsc,
    const int* __restrict__ bi_idx, const float* __restrict__ logits,
    const float* __restrict__ colmax, const float* __restrict__ colsum,
    int baseRow, float* __restrict__ partials) {
    const int lane = threadIdx.x & 31;
    const int wave = threadIdx.x >> 5;
    const int r    = blockIdx.x * 8 + wave;
    const int gr   = baseRow + r;

    float mu[4], ls[4];
#pragma unroll
    for (int i = 0; i < 4; ++i) {
        int k = lane + 32 * i;
        mu[i] = muc[(size_t)r * PK + k];
        ls[i] = lsc[(size_t)r * PK + k];
    }
    float kld = 0.0f, mx = -3.4e38f;
#pragma unroll
    for (int i = 0; i < 4; ++i) {
        kld += 1.0f + ls[i] - mu[i] * mu[i] - expf(ls[i]);
        mx = fmaxf(mx, mu[i]);
    }
#pragma unroll
    for (int off = 16; off > 0; off >>= 1) {
        kld += __shfl_xor(kld, off, 32);
        mx   = fmaxf(mx, __shfl_xor(mx, off, 32));
    }
    float e[4], den = 0.0f;
#pragma unroll
    for (int i = 0; i < 4; ++i) { e[i] = expf(mu[i] - mx); den += e[i]; }
#pragma unroll
    for (int off = 16; off > 0; off >>= 1) den += __shfl_xor(den, off, 32);

    const int i0 = bi_idx[2 * gr];
    const int i1 = bi_idx[2 * gr + 1];
    float res = 0.0f;
#pragma unroll
    for (int i = 0; i < 4; ++i) {
        int k = lane + 32 * i;
        float t0 = expf(logits[(size_t)i0 * PK + k] - colmax[k]) / colsum[k];
        float t1 = expf(logits[(size_t)i1 * PK + k] - colmax[k]) / colsum[k];
        float th = e[i] / den;
        res += th * th * t0 * t1;
    }
#pragma unroll
    for (int off = 16; off > 0; off >>= 1) res += __shfl_xor(res, off, 32);

    __shared__ float s0[8], s1[8];
    if (lane == 0) { s0[wave] = logf(res + 1e-6f); s1[wave] = kld; }
    __syncthreads();
    if (threadIdx.x == 0) {
        float a = 0.0f, b = 0.0f;
#pragma unroll
        for (int w = 0; w < 8; ++w) { a += s0[w]; b += s1[w]; }
        partials[(size_t)blockIdx.x * 2 + 0] = a;
        partials[(size_t)blockIdx.x * 2 + 1] = b;
    }
}

// ---------------- deterministic final reduction ----------------
__global__ __launch_bounds__(256) void final_reduce(const float* __restrict__ partials,
                                                    int nblocks, float* __restrict__ out) {
    __shared__ float r0[256], r1[256];
    float a = 0.0f, b = 0.0f;
    for (int i = threadIdx.x; i < nblocks; i += 256) {
        a += partials[2 * i];
        b += partials[2 * i + 1];
    }
    r0[threadIdx.x] = a; r1[threadIdx.x] = b; __syncthreads();
    for (int s = 128; s > 0; s >>= 1) {
        if (threadIdx.x < s) { r0[threadIdx.x] += r0[threadIdx.x + s]; r1[threadIdx.x] += r1[threadIdx.x + s]; }
        __syncthreads();
    }
    if (threadIdx.x == 0) {
        const float invB = 1.0f / (float)PB;
        out[0] = r0[0] * invB;          // recon_loss
        out[1] = -0.5f * r1[0] * invB;  // kld
    }
}

// ---------------- launch ----------------
static inline size_t alignUp(size_t x) { return (x + 255) & ~(size_t)255; }

extern "C" void kernel_launch(void* const* d_in, const int* in_sizes, int n_in,
                              void* d_out, int out_size, void* d_ws, size_t ws_size,
                              hipStream_t stream) {
    (void)in_sizes; (void)n_in; (void)out_size; (void)ws_size;
    const int*   bi_idx   = (const int*)  d_in[0];
    const float* biterms  = (const float*)d_in[1];
    const float* rho      = (const float*)d_in[2];
    const float* alphas_W = (const float*)d_in[3];
    const float* q1_W = (const float*)d_in[4];
    const float* q1_b = (const float*)d_in[5];
    const float* q2_W = (const float*)d_in[6];
    const float* q2_b = (const float*)d_in[7];
    const float* mu_W = (const float*)d_in[8];
    const float* mu_b = (const float*)d_in[9];
    const float* ls_W = (const float*)d_in[10];
    const float* ls_b = (const float*)d_in[11];

    // workspace carve (padded shapes so all fragment loads are branchless)
    char* ws = (char*)d_ws;
    size_t off = 0;
    auto carve = [&](size_t bytes) { char* p = ws + off; off += alignUp(bytes); return p; };
    _Float16* alphast = (_Float16*)carve((size_t)PK * KPAD_E * 2);      // [128][320]
    _Float16* q1t     = (_Float16*)carve((size_t)PHPAD * KPAD_E * 2);   // [832][320]
    _Float16* q2t     = (_Float16*)carve((size_t)PHPAD * PH * 2);       // [832][800]
    _Float16* mut     = (_Float16*)carve((size_t)PK * PH * 2);          // [128][800]
    _Float16* lst     = (_Float16*)carve((size_t)PK * PH * 2);          // [128][800]
    _Float16* rhoh    = (_Float16*)carve((size_t)PVPAD * KPAD_E * 2);   // [50048][320]
    float*    logits  = (float*)   carve((size_t)PV * PK * 4);
    float*    colmax  = (float*)   carve((size_t)PK * 4);
    float*    colsum  = (float*)   carve((size_t)PK * 4);
    _Float16* bitc    = (_Float16*)carve((size_t)BCHUNK * KPAD_E * 2);
    _Float16* h1c     = (_Float16*)carve((size_t)BCHUNK * PH * 2);
    _Float16* h2c     = (_Float16*)carve((size_t)BCHUNK * PH * 2);
    float*    muc     = (float*)   carve((size_t)BCHUNK * PK * 4);
    float*    lsc     = (float*)   carve((size_t)BCHUNK * PK * 4);
    float*    partials= (float*)   carve((size_t)(PB / 8) * 2 * 4);

    // --- weight prep (transpose + f16, zero-pad K and N) ---
    auto prep = [&](const float* in, _Float16* out, int Kreal, int Nreal, int Kpad, int Npad) {
        int total = Npad * Kpad;
        prep_transpose<<<(total + 255) / 256, 256, 0, stream>>>(in, out, Kreal, Nreal, Kpad, Npad);
    };
    prep(alphas_W, alphast, PE, PK, KPAD_E, PK);
    prep(q1_W,     q1t,     PE, PH, KPAD_E, PHPAD);
    prep(q2_W,     q2t,     PH, PH, PH,     PHPAD);
    prep(mu_W,     mut,     PH, PK, PH,     PK);
    prep(ls_W,     lst,     PH, PK, PH,     PK);

    // --- rho -> padded f16 (rows padded to 50048 with zeros) ---
    {
        long total = (long)PVPAD * KPAD_E;
        pad_convert<<<(int)((total + 255) / 256), 256, 0, stream>>>(rho, rhoh, PE, KPAD_E, PV, total);
    }

    // --- beta logits + column softmax stats ---
    {
        dim3 g(PVPAD / 128, (PK + 63) / 64);
        gemm_kernel<false, false><<<g, 256, 0, stream>>>(
            rhoh, KPAD_E, alphast, KPAD_E, nullptr,
            logits, nullptr, PK, PV, PK);
        col_stats<<<PK, 256, 0, stream>>>(logits, colmax, colsum);
    }

    // --- MLP in batch chunks ---
    for (int c = 0; c < NCHUNK; ++c) {
        const int base = c * BCHUNK;
        {
            long total = (long)BCHUNK * KPAD_E;
            pad_convert<<<(int)((total + 255) / 256), 256, 0, stream>>>(
                biterms + (size_t)base * PE, bitc, PE, KPAD_E, BCHUNK, total);
        }
        dim3 g1(BCHUNK / 128, (PH + 63) / 64);
        gemm_kernel<true, true><<<g1, 256, 0, stream>>>(
            bitc, KPAD_E, q1t, KPAD_E, q1_b,
            nullptr, h1c, PH, BCHUNK, PH);
        gemm_kernel<true, true><<<g1, 256, 0, stream>>>(
            h1c, PH, q2t, PH, q2_b,
            nullptr, h2c, PH, BCHUNK, PH);
        dim3 g3(BCHUNK / 128, (PK + 63) / 64);
        gemm_kernel<false, false><<<g3, 256, 0, stream>>>(
            h2c, PH, mut, PH, mu_b,
            muc, nullptr, PK, BCHUNK, PK);
        gemm_kernel<false, false><<<g3, 256, 0, stream>>>(
            h2c, PH, lst, PH, ls_b,
            lsc, nullptr, PK, BCHUNK, PK);
        finalize_rows<<<BCHUNK / 8, 256, 0, stream>>>(
            muc, lsc, bi_idx, logits, colmax, colsum, base,
            partials + (size_t)(base / 8) * 2);
    }

    final_reduce<<<1, 256, 0, stream>>>(partials, PB / 8, (float*)d_out);
}
